// RNASeqHyena_2104533975838
// MI455X (gfx1250) — matrix operand order repaired and checked by reading, not compiled, not run on hardware
//
#include <hip/hip_runtime.h>
#include <cstdint>
#include <cstddef>
#include <cstring>

typedef _Float16 h16;
typedef __attribute__((ext_vector_type(16))) _Float16 v16h;
typedef __attribute__((ext_vector_type(8)))  _Float16 v8h;
typedef __attribute__((ext_vector_type(8)))  float    v8f;

#define B_ 16
#define L_ 2048
#define LP_ (L_ + 32)   // reversed filter row stride incl. zero pad
#define D_ 256
#define CHUNK_ 10
#define INPUT_DIM_ 20480
#define FF_ 1024
#define NLAYERS_ 4
#define FILT_EMB_ 16
#define FILT_HID_ 64
#define NCLS_ 10
#define EPS_ 1e-5f
#define LOG1E4 9.210340371976184f

__device__ __forceinline__ float gelu_f(float x) {
  return 0.5f * x * (1.f + erff(x * 0.7071067811865475f));
}

__device__ __forceinline__ v16h make16(v8h lo, v8h hi) {
  v16h r;
#pragma unroll
  for (int e = 0; e < 8; ++e) { r[e] = lo[e]; r[e + 8] = hi[e]; }
  return r;
}

__device__ __forceinline__ float block_reduce_256(float v, float* red) {
  int tid = threadIdx.x;
  red[tid] = v;
  __syncthreads();
  for (int s = 128; s > 0; s >>= 1) {
    if (tid < s) red[tid] += red[tid + s];
    __syncthreads();
  }
  float r = red[0];
  __syncthreads();
  return r;
}

// ---------------- embed: xs = x.reshape(B,L,10) @ inp_w + inp_b + PE(L,256) -----------
__global__ void k_embed(const float* __restrict__ x, const float* __restrict__ w,
                        const float* __restrict__ bias, float* __restrict__ XS) {
  __shared__ float xr[CHUNK_];
  int tok = blockIdx.x, d = threadIdx.x;
  int b = tok >> 11, t = tok & (L_ - 1);
  if (d < CHUNK_) xr[d] = x[(size_t)b * INPUT_DIM_ + t * CHUNK_ + d];
  __syncthreads();
  float s = bias[d];
#pragma unroll
  for (int c = 0; c < CHUNK_; ++c) s += xr[c] * w[c * D_ + d];
  int m = d >> 1;
  float ang = (float)t * expf(-(float)(2 * m) * (LOG1E4 / (float)D_));
  s += (d & 1) ? cosf(ang) : sinf(ang);
  XS[(size_t)tok * D_ + d] = s;
}

// -------- implicit filter, stored REVERSED + zero-padded: HR[d][u] = h[L-1-u][d] ------
// HR[d][L..L+31] = 0 so the causal-conv Toeplitz fragment never needs masking.
__global__ void k_filter(const float* __restrict__ fw1, const float* __restrict__ fb1,
                         const float* __restrict__ fw2, const float* __restrict__ fb2,
                         h16* __restrict__ HR) {
  __shared__ float pe[FILT_EMB_];
  __shared__ float hid[FILT_HID_];
  int t = blockIdx.x, d = threadIdx.x;
  if (d < FILT_EMB_) {
    int m = d >> 1;
    float ang = (float)t * expf(-(float)(2 * m) * (LOG1E4 / (float)FILT_EMB_));
    pe[d] = (d & 1) ? cosf(ang) : sinf(ang);
  }
  __syncthreads();
  if (d < FILT_HID_) {
    float s = fb1[d];
#pragma unroll
    for (int k = 0; k < FILT_EMB_; ++k) s += pe[k] * fw1[k * FILT_HID_ + d];
    hid[d] = gelu_f(s);
  }
  __syncthreads();
  float s = fb2[d];
#pragma unroll
  for (int j = 0; j < FILT_HID_; ++j) s += hid[j] * fw2[j * D_ + d];
  HR[(size_t)d * LP_ + (L_ - 1 - t)] = (h16)s;
  if (t < 32) HR[(size_t)d * LP_ + L_ + t] = (h16)0.f;  // zero pad
}

// ---------------- layernorm over D=256, output f16 row-major --------------------------
__global__ void k_ln(const float* __restrict__ X, const float* __restrict__ g,
                     const float* __restrict__ b, h16* __restrict__ O) {
  __shared__ float red[256];
  int tok = blockIdx.x, d = threadIdx.x;
  size_t base = (size_t)tok * D_;
  float v = X[base + d];
  float mu = block_reduce_256(v, red) * (1.f / D_);
  float c = v - mu;
  float var = block_reduce_256(c * c, red) * (1.f / D_);
  O[base + d] = (h16)(c * rsqrtf(var + EPS_) * g[d] + b[d]);
}

// ---------------- f32 [K][N] weight -> f16 transposed [N][K] --------------------------
__global__ void k_castT(const float* __restrict__ W, h16* __restrict__ O, int K, int N) {
  int i = blockIdx.x * 256 + threadIdx.x;
  if (i >= K * N) return;
  int n = i / K, k = i - n * K;
  O[i] = (h16)W[(size_t)k * N + n];
}

// ---------------- WMMA GEMM: C[M,N] = A[M,K](f16) * Bt[N,K](f16) + bias ---------------
#define BM 128
#define BN 64
#define BK 32
#define LDT 40  // LDS row stride in halfs (80B: 16B-aligned, bank-skewed)

template <bool GELU, bool RESID, bool WF32, bool WF16>
__global__ void __launch_bounds__(256)
k_gemm(const h16* __restrict__ A, const h16* __restrict__ Bt,
       const float* __restrict__ bias, float* __restrict__ Cf,
       h16* __restrict__ Ch, int M, int N, int K) {
  __shared__ alignas(16) h16 As[BM * LDT];
  __shared__ alignas(16) h16 Bs[BN * LDT];
  const int tid = threadIdx.x;
  const int lane = tid & 31, wid = tid >> 5;
  const int wm = wid & 3, wn = wid >> 2;      // 4x2 wave grid, 32x32 per wave
  const int lr = lane & 15, lh = lane >> 4;
  const int klo = lh * 8;
  const int nblk = N / BN;
  const int m0 = (blockIdx.x / nblk) * BM;
  const int n0 = (blockIdx.x % nblk) * BN;

  v8f acc[2][2] = {};
  for (int k0 = 0; k0 < K; k0 += BK) {
    __syncthreads();
    for (int c = tid; c < BM * 4; c += 256) {
      int r = c >> 2, cc = (c & 3) << 3;
      *(v8h*)&As[r * LDT + cc] = *(const v8h*)&A[(size_t)(m0 + r) * K + k0 + cc];
    }
    for (int c = tid; c < BN * 4; c += 256) {
      int r = c >> 2, cc = (c & 3) << 3;
      *(v8h*)&Bs[r * LDT + cc] = *(const v8h*)&Bt[(size_t)(n0 + r) * K + k0 + cc];
    }
    __syncthreads();
    v16h af[2], bf[2];
#pragma unroll
    for (int i = 0; i < 2; ++i) {
      const h16* p = &As[(wm * 32 + i * 16 + lr) * LDT];
      af[i] = make16(*(const v8h*)&p[klo], *(const v8h*)&p[16 + klo]);
    }
#pragma unroll
    for (int j = 0; j < 2; ++j) {
      const h16* p = &Bs[(wn * 32 + j * 16 + lr) * LDT];
      bf[j] = make16(*(const v8h*)&p[klo], *(const v8h*)&p[16 + klo]);
    }
#pragma unroll
    for (int i = 0; i < 2; ++i)
#pragma unroll
      for (int j = 0; j < 2; ++j)
        acc[i][j] = __builtin_amdgcn_wmma_f32_16x16x32_f16(
            false, af[i], false, bf[j], (short)0, acc[i][j], false, false);
  }
#pragma unroll
  for (int i = 0; i < 2; ++i) {
#pragma unroll
    for (int j = 0; j < 2; ++j) {
      int n = n0 + wn * 32 + j * 16 + lr;
      int mb = m0 + wm * 32 + i * 16 + lh * 8;
      float bn = bias[n];
#pragma unroll
      for (int r = 0; r < 8; ++r) {
        size_t idx = (size_t)(mb + r) * N + n;
        float v = acc[i][j][r] + bn;
        if (GELU) v = gelu_f(v);
        if (RESID) v += Cf[idx];
        if (WF32) Cf[idx] = v;
        if (WF16) Ch[idx] = (h16)v;
      }
    }
  }
}

// ---------------- depthwise short conv (k=3, same pad) -> x0c f16 [D][B][L] ----------
__global__ void k_shortconv(const float* __restrict__ P, const float* __restrict__ sw,
                            const float* __restrict__ sb, h16* __restrict__ XC) {
  int b = blockIdx.x >> 4;
  int t0 = (blockIdx.x & 15) << 7;
  int d = threadIdx.x;
  float w0 = sw[d * 3 + 0], w1 = sw[d * 3 + 1], w2 = sw[d * 3 + 2], bb = sb[d];
  for (int tt = 0; tt < 128; ++tt) {
    int t = t0 + tt;
    size_t row = (size_t)(b * L_ + t) * (3 * D_);
    float xm = (t > 0) ? P[row - 3 * D_ + d] : 0.f;
    float xc = P[row + d];
    float xp = (t < L_ - 1) ? P[row + 3 * D_ + d] : 0.f;
    XC[(size_t)d * (B_ * L_) + (size_t)b * L_ + t] = (h16)(w0 * xm + w1 * xc + w2 * xp + bb);
  }
}

// ------- causal long conv via blocked Toeplitz WMMA: Y[b,t,d] = sum_{s<=t} X*h(t-s) ---
// Toeplitz fragment T[k][n] = h[(t0+n)-(s0+k)] is read branch-free from the reversed,
// zero-padded filter HR: h[top-e] == HR[(L-1-top)+e], non-causal entries hit the pad.
__global__ void __launch_bounds__(128)
k_conv(const h16* __restrict__ X, const h16* __restrict__ HR, float* __restrict__ Y) {
  int gw = blockIdx.x * 4 + (threadIdx.x >> 5);  // one wave per (d, t-tile)
  int lane = threadIdx.x & 31;
  int lr = lane & 15, lh = lane >> 4;
  int d = gw >> 7;             // 128 t-tiles per channel
  int t0 = (gw & 127) << 4;
  const h16* xrow = X + (size_t)d * (B_ * L_) + (size_t)lr * L_;  // batch lr = M row
  // base index into reversed filter for s0 = 0: (L-1) - (t0 + lr - 16*lh)
  const h16* hrev = HR + (size_t)d * LP_ + (L_ - 1 - t0 - lr + (lh << 4));
  const int klo = lh * 8;
  v8f acc = {};
  for (int s0 = 0; s0 <= t0; s0 += 32) {
    // A fragment: X[b=0..15, s0..s0+31]  (two aligned 16B loads per lane)
    v16h af = make16(*(const v8h*)&xrow[s0 + (lh << 4)], *(const v8h*)&xrow[s0 + (lh << 4) + 8]);
    // B fragment: 16 contiguous halfs from reversed filter (unaligned OK, no branches)
    v8h b0, b1;
    __builtin_memcpy(&b0, hrev + s0, 16);
    __builtin_memcpy(&b1, hrev + s0 + 8, 16);
    v16h bf = make16(b0, b1);
    acc = __builtin_amdgcn_wmma_f32_16x16x32_f16(false, af, false, bf, (short)0, acc,
                                                 false, false);
  }
#pragma unroll
  for (int r = 0; r < 8; ++r) {
    int b = r + lh * 8;
    Y[((size_t)b * L_ + (t0 + lr)) * D_ + d] = acc[r];
  }
}

// ---------------- gating: g = y * x1 * x2 -> f16 --------------------------------------
__global__ void k_gate(const float* __restrict__ Y, const float* __restrict__ P,
                       h16* __restrict__ G) {
  size_t i = (size_t)blockIdx.x * 256 + threadIdx.x;
  size_t tok = i >> 8;
  int d = (int)(i & 255);
  size_t pr = tok * (3 * D_);
  G[i] = (h16)(Y[i] * P[pr + D_ + d] * P[pr + 2 * D_ + d]);
}

// ---------------- head: mean over L, layernorm, cls -----------------------------------
__global__ void k_head(const float* __restrict__ XS, const float* __restrict__ g,
                       const float* __restrict__ bb, const float* __restrict__ cw,
                       const float* __restrict__ cb, float* __restrict__ out) {
  __shared__ float red[256];
  __shared__ float nrm[256];
  int b = blockIdx.x, d = threadIdx.x;
  float s = 0.f;
  const float* base = XS + (size_t)b * L_ * D_ + d;
  for (int t = 0; t < L_; ++t) s += base[(size_t)t * D_];
  float p = s * (1.f / L_);
  float mu = block_reduce_256(p, red) * (1.f / D_);
  float c = p - mu;
  float var = block_reduce_256(c * c, red) * (1.f / D_);
  nrm[d] = c * rsqrtf(var + EPS_) * g[d] + bb[d];
  __syncthreads();
  if (d < NCLS_) {
    float o = cb[d];
    for (int k = 0; k < D_; ++k) o += nrm[k] * cw[k * NCLS_ + d];
    out[b * NCLS_ + d] = o;
  }
}

extern "C" void kernel_launch(void* const* d_in, const int* in_sizes, int n_in,
                              void* d_out, int out_size, void* d_ws, size_t ws_size,
                              hipStream_t stream) {
  (void)in_sizes; (void)n_in; (void)out_size; (void)ws_size;
  const float* x         = (const float*)d_in[0];
  const float* inp_w     = (const float*)d_in[1];
  const float* inp_b     = (const float*)d_in[2];
  const float* in_proj_w = (const float*)d_in[3];
  const float* in_proj_b = (const float*)d_in[4];
  const float* short_w   = (const float*)d_in[5];
  const float* short_b   = (const float*)d_in[6];
  const float* filt_w1   = (const float*)d_in[7];
  const float* filt_b1   = (const float*)d_in[8];
  const float* filt_w2   = (const float*)d_in[9];
  const float* filt_b2   = (const float*)d_in[10];
  const float* out_w     = (const float*)d_in[11];
  const float* out_b     = (const float*)d_in[12];
  const float* ln1_g     = (const float*)d_in[13];
  const float* ln1_b     = (const float*)d_in[14];
  const float* ln2_g     = (const float*)d_in[15];
  const float* ln2_b     = (const float*)d_in[16];
  const float* ffn_w1    = (const float*)d_in[17];
  const float* ffn_b1    = (const float*)d_in[18];
  const float* ffn_w2    = (const float*)d_in[19];
  const float* ffn_b2    = (const float*)d_in[20];
  const float* norm_g    = (const float*)d_in[21];
  const float* norm_b    = (const float*)d_in[22];
  const float* cls_w     = (const float*)d_in[23];
  const float* cls_b     = (const float*)d_in[24];
  float* outp = (float*)d_out;

  char* ws = (char*)d_ws;
  size_t off = 0;
  auto alloc = [&](size_t bytes) {
    char* p = ws + off;
    off = (off + bytes + 255) & ~(size_t)255;
    return p;
  };
  const size_t NT = (size_t)B_ * L_;  // 32768 tokens
  float* xs   = (float*)alloc(NT * D_ * 4);        // residual stream (B,L,D)
  float* proj = (float*)alloc(NT * 3 * D_ * 4);    // x0|x1|x2
  float* yv   = (float*)alloc(NT * D_ * 4);        // conv output (B,L,D)
  h16* ah     = (h16*)alloc(NT * D_ * 2);          // f16 GEMM-A staging
  h16* ffnh   = (h16*)alloc(NT * FF_ * 2);         // gelu(ffn1) f16
  h16* x0c    = (h16*)alloc(NT * D_ * 2);          // short-conv out, [D][B][L]
  h16* hr     = (h16*)alloc((size_t)D_ * LP_ * 2); // reversed+padded filter, [D][L+32]
  h16* w_inp  = (h16*)alloc((size_t)D_ * 3 * D_ * 2);
  h16* w_out  = (h16*)alloc((size_t)D_ * D_ * 2);
  h16* w_f1   = (h16*)alloc((size_t)D_ * FF_ * 2);
  h16* w_f2   = (h16*)alloc((size_t)FF_ * D_ * 2);

  k_embed<<<(unsigned)NT, 256, 0, stream>>>(x, inp_w, inp_b, xs);

  for (int i = 0; i < NLAYERS_; ++i) {
    k_castT<<<(D_ * 3 * D_ + 255) / 256, 256, 0, stream>>>(
        in_proj_w + (size_t)i * D_ * 3 * D_, w_inp, D_, 3 * D_);
    k_castT<<<(D_ * D_ + 255) / 256, 256, 0, stream>>>(
        out_w + (size_t)i * D_ * D_, w_out, D_, D_);
    k_castT<<<(D_ * FF_ + 255) / 256, 256, 0, stream>>>(
        ffn_w1 + (size_t)i * D_ * FF_, w_f1, D_, FF_);
    k_castT<<<(FF_ * D_ + 255) / 256, 256, 0, stream>>>(
        ffn_w2 + (size_t)i * FF_ * D_, w_f2, FF_, D_);
    k_filter<<<L_, 256, 0, stream>>>(filt_w1 + i * FILT_EMB_ * FILT_HID_,
                                     filt_b1 + i * FILT_HID_,
                                     filt_w2 + i * FILT_HID_ * D_, filt_b2 + i * D_, hr);
    // ln1 -> f16, in_proj GEMM
    k_ln<<<(unsigned)NT, 256, 0, stream>>>(xs, ln1_g + i * D_, ln1_b + i * D_, ah);
    k_gemm<false, false, true, false><<<(NT / BM) * (3 * D_ / BN), 256, 0, stream>>>(
        ah, w_inp, in_proj_b + i * 3 * D_, proj, nullptr, (int)NT, 3 * D_, D_);
    // short conv, long causal conv (Toeplitz WMMA), gating
    k_shortconv<<<B_ * 16, 256, 0, stream>>>(proj, short_w + i * D_ * 3,
                                             short_b + i * D_, x0c);
    k_conv<<<D_ * (L_ / 16) / 4, 128, 0, stream>>>(x0c, hr, yv);
    k_gate<<<(unsigned)(NT * D_ / 256), 256, 0, stream>>>(yv, proj, ah);
    // out proj + residual into xs
    k_gemm<false, true, true, false><<<(NT / BM) * (D_ / BN), 256, 0, stream>>>(
        ah, w_out, out_b + i * D_, xs, nullptr, (int)NT, D_, D_);
    // ffn: ln2 -> gelu(GEMM) -> GEMM + residual
    k_ln<<<(unsigned)NT, 256, 0, stream>>>(xs, ln2_g + i * D_, ln2_b + i * D_, ah);
    k_gemm<true, false, false, true><<<(NT / BM) * (FF_ / BN), 256, 0, stream>>>(
        ah, w_f1, ffn_b1 + i * FF_, nullptr, ffnh, (int)NT, FF_, D_);
    k_gemm<false, true, true, false><<<(NT / BM) * (D_ / BN), 256, 0, stream>>>(
        ffnh, w_f2, ffn_b2 + i * D_, xs, nullptr, (int)NT, D_, FF_);
  }
  k_head<<<B_, 256, 0, stream>>>(xs, norm_g, norm_b, cls_w, cls_b, outp);
}
